// QuantLinear_33844342292936
// MI455X (gfx1250) — 8-bit matrix operand order repaired and checked by reading, compiled, not run on hardware
//
#include <hip/hip_runtime.h>
#include <stdint.h>
#include <stddef.h>

typedef int v8i __attribute__((ext_vector_type(8)));

#define N_ROWS 16384
#define D_IN   1024
#define D_OUT  1024


__global__ void k_init(unsigned* ws) {
    if (threadIdx.x < 2) ws[threadIdx.x] = 0u;
}

__global__ __launch_bounds__(256) void k_absmax(const float* __restrict__ src,
                                                int n, unsigned* __restrict__ out) {
    __shared__ unsigned red[256];
    unsigned m = 0u;
    for (int i = blockIdx.x * blockDim.x + threadIdx.x; i < n;
         i += gridDim.x * blockDim.x) {
        unsigned b = __float_as_uint(fabsf(src[i]));
        m = m > b ? m : b;
    }
    red[threadIdx.x] = m;
    __syncthreads();
    for (int s = 128; s > 0; s >>= 1) {
        if (threadIdx.x < (unsigned)s) {
            unsigned o = red[threadIdx.x + s];
            if (o > red[threadIdx.x]) red[threadIdx.x] = o;
        }
        __syncthreads();
    }
    if (threadIdx.x == 0) atomicMax(out, red[0]);
}

__device__ __forceinline__ int kbaseA(int v, int half) {
    return ((v >= 4) ? 32 : 0) + ((v & 3) >> 1) * 16 + (v & 1) * 4 + half * 8;
}

__device__ __forceinline__ int kbaseB(int v, int half) {
    return ((v >= 4) ? 32 : 0) + ((v & 3) >> 1) * 16 + (v & 1) * 4 + half * 8;
}

__device__ __forceinline__ int quant4(const float* __restrict__ s, float inv) {
    int out = 0;
#pragma unroll
    for (int b = 0; b < 4; ++b) {
        float q = rintf(s[b] * inv);
        q = fminf(127.0f, fmaxf(-127.0f, q));
        out |= ((int)q & 0xff) << (8 * b);
    }
    return out;
}

__global__ __launch_bounds__(256) void k_quant_a(const float* __restrict__ x,
                                                 const unsigned* __restrict__ amax,
                                                 int* __restrict__ dst) {
    int d = blockIdx.x * blockDim.x + threadIdx.x;
    int tile = d >> 8, lane = (d >> 3) & 31, v = d & 7;
    int mt = tile >> 4, kt = tile & 15;
    int m = mt * 16 + (lane & 15);
    int k = kt * 64 + kbaseA(v, lane >> 4);
    float inv = 127.0f / __uint_as_float(*amax);
    dst[d] = quant4(x + (size_t)m * D_IN + k, inv);
}

__global__ __launch_bounds__(256) void k_quant_b(const float* __restrict__ w,
                                                 const unsigned* __restrict__ amax,
                                                 int* __restrict__ dst) {
    int d = blockIdx.x * blockDim.x + threadIdx.x;
    int tile = d >> 8, lane = (d >> 3) & 31, v = d & 7;
    int nt = tile >> 4, kt = tile & 15;
    int n = nt * 16 + (lane & 15);
    int k = kt * 64 + kbaseB(v, lane >> 4);
    float inv = 127.0f / __uint_as_float(*amax);
    dst[d] = quant4(w + (size_t)n * D_IN + k, inv);
}

__global__ __launch_bounds__(256) void k_gemm(const char* __restrict__ qa,
                                              const char* __restrict__ qb,
                                              const unsigned* __restrict__ amax,
                                              const float* __restrict__ bias,
                                              float* __restrict__ y) {
    const int lane = threadIdx.x & 31;
    const int w    = blockIdx.x * 8 + (threadIdx.x >> 5);
    const int wm   = w >> 4;
    const int wn   = w & 15;
    const int laneOff = lane * 32;

    v8i acc[4][4];
#pragma unroll
    for (int i = 0; i < 4; ++i)
#pragma unroll
        for (int j = 0; j < 4; ++j) acc[i][j] = (v8i)(0);

    for (int kt = 0; kt < 16; ++kt) {
        v8i b[4];
#pragma unroll
        for (int j = 0; j < 4; ++j)
            b[j] = *(const v8i*)(qb + (size_t)(((wn * 4 + j) * 16 + kt) * 1024) + laneOff);
#pragma unroll
        for (int i = 0; i < 4; ++i) {
            v8i a = *(const v8i*)(qa + (size_t)(((wm * 4 + i) * 16 + kt)) * 1024 + laneOff);
#pragma unroll
            for (int j = 0; j < 4; ++j) {
                acc[i][j] = __builtin_amdgcn_wmma_i32_16x16x64_iu8(
                    true, a, true, b[j], acc[i][j], false, false);
            }
        }
    }

    const float sx = __uint_as_float(amax[0]) * (1.0f / 127.0f);
    const float sw = __uint_as_float(amax[1]) * (1.0f / 127.0f);
    const float sc = sx * sw;

    const int half = lane >> 4, nloc = lane & 15;
    float bcol[4];
#pragma unroll
    for (int j = 0; j < 4; ++j)
        bcol[j] = bias[(wn * 4 + j) * 16 + nloc];

#pragma unroll
    for (int i = 0; i < 4; ++i) {
        const int mbase = (wm * 4 + i) * 16 + half * 8;
#pragma unroll
        for (int j = 0; j < 4; ++j) {
            const int nbase = (wn * 4 + j) * 16 + nloc;
#pragma unroll
            for (int r = 0; r < 8; ++r) {
                y[(size_t)(mbase + r) * D_OUT + nbase] =
                    (float)acc[i][j][r] * sc + bcol[j];
            }
        }
    }
}

extern "C" void kernel_launch(void* const* d_in, const int* in_sizes, int n_in,
                              void* d_out, int out_size, void* d_ws, size_t ws_size,
                              hipStream_t stream) {
    (void)in_sizes; (void)n_in; (void)out_size; (void)ws_size;

    const float* x    = (const float*)d_in[0];
    const float* wgt  = (const float*)d_in[1];
    const float* bias = (const float*)d_in[2];
    float* y          = (float*)d_out;

    unsigned* amax = (unsigned*)d_ws;
    char* qa = (char*)d_ws + 256;
    char* qb = qa + (size_t)N_ROWS * D_IN;

    k_init<<<1, 32, 0, stream>>>(amax);
    k_absmax<<<1024, 256, 0, stream>>>(x,   N_ROWS * D_IN, amax + 0);
    k_absmax<<<256,  256, 0, stream>>>(wgt, D_OUT * D_IN, amax + 1);
    k_quant_a<<<(N_ROWS * D_IN / 4) / 256, 256, 0, stream>>>(x,   amax + 0, (int*)qa);
    k_quant_b<<<(D_OUT  * D_IN / 4) / 256, 256, 0, stream>>>(wgt, amax + 1, (int*)qb);
    k_gemm<<<(N_ROWS / 64) * (D_OUT / 64) / 8, 256, 0, stream>>>(qa, qb, amax, bias, y);
}
